// SogCLR_Loss_22643067584620
// MI455X (gfx1250) — compile-verified, matrix operand order checked by reading, not run on hardware
//
#include <hip/hip_runtime.h>
#include <hip/hip_bf16.h>

// ---------------------------------------------------------------------------
// SogCLR loss on MI455X (gfx1250, wave32, WMMA).
//   B = 4096, D = 1024, T = 0.1, GAMMA = 0.8, ALPHA = 0.5, EPS = 1e-14
// Pipeline: normalize(f32->f16) -> WMMA f16 GEMM (f32 acc, async double-
//           buffered LDS) -> row pass -> col pass -> final reduction.
// ---------------------------------------------------------------------------

typedef __attribute__((ext_vector_type(16))) _Float16 v16h;
typedef __attribute__((ext_vector_type(8)))  _Float16 h8v;
typedef __attribute__((ext_vector_type(8)))  float    v8f;
typedef __attribute__((ext_vector_type(4)))  int      v4i_t;

#define INV_T   10.0f
#define GAMMA_C 0.8f
#define ALPHA_C 0.5f
#define EPS_C   1e-14f

#define LDST 40   // LDS row stride in halves (80B) -> conflict-free b128 reads

#if __has_builtin(__builtin_amdgcn_global_load_async_to_lds_b128)
#define HAVE_ASYNC_LDS 1
typedef __attribute__((address_space(1))) v4i_t glb_v4i;
typedef __attribute__((address_space(3))) v4i_t lds_v4i;
#endif

static __device__ __forceinline__ void wait_async_all() {
#if __has_builtin(__builtin_amdgcn_s_wait_asynccnt)
    __builtin_amdgcn_s_wait_asynccnt(0);
#else
    asm volatile("s_wait_asynccnt 0x0" ::: "memory");
#endif
}

// ---------------------------------------------------------------------------
// Kernel 1: L2-normalize rows of zis and zjs, emit f16.
// grid = 2*B blocks, 256 threads. rows [0,B) -> zis/zn, rows [B,2B) -> zjs/wn.
// ---------------------------------------------------------------------------
__global__ __launch_bounds__(256) void normalize_rows(
    const float* __restrict__ zis, const float* __restrict__ zjs,
    _Float16* __restrict__ zn, _Float16* __restrict__ wn, int Bn, int D)
{
    int row = blockIdx.x;
    const float* src;
    _Float16*    dst;
    if (row < Bn) { src = zis + (size_t)row * D;        dst = zn + (size_t)row * D; }
    else          { src = zjs + (size_t)(row - Bn) * D; dst = wn + (size_t)(row - Bn) * D; }

    __shared__ float red[256];
    int tid = threadIdx.x;
    float ss = 0.0f;
    for (int k = tid; k < D; k += 256) { float v = src[k]; ss = fmaf(v, v, ss); }
    red[tid] = ss; __syncthreads();
    for (int s = 128; s > 0; s >>= 1) {
        if (tid < s) red[tid] += red[tid + s];
        __syncthreads();
    }
    float rn = rsqrtf(red[0]);
    for (int k = tid; k < D; k += 256) dst[k] = (_Float16)(src[k] * rn);
}

// ---------------------------------------------------------------------------
// Kernel 2: sim = zn * wn^T via v_wmma_f32_16x16x32_f16.
// 256 threads (8 waves) per block compute a 128x128 tile. Wave (wM,wN) with
// wM in 0..3, wN in 0..1 owns a 32x64 strip: 2 (M) x 4 (N) accumulators.
// K staged in steps of 32 through double-buffered LDS; next tile is fetched
// with GLOBAL_LOAD_ASYNC_TO_LDS_B128 (ASYNCcnt) while WMMAs consume the
// current one. Padded stride 40 halves -> conflict-free ds_load_b128 frags.
//
// ISA fragment layout (16-bit A 16x32 / B 32x16, wave32):
//   lane L: m/n = L&15, kh = L>>4
//   elems 0..7  <- K = kh*8 + 0..7        (one contiguous 16B chunk)
//   elems 8..15 <- K = 16 + kh*8 + 0..7   (second contiguous 16B chunk)
// C/D f32 layout: VGPR r -> M = r + 8*(L>>4), N = L&15.
// ---------------------------------------------------------------------------
union HV16 { v16h v; h8v h[2]; };

__global__ __launch_bounds__(256) void gemm_wmma(
    const _Float16* __restrict__ A, const _Float16* __restrict__ Bm,
    float* __restrict__ C, int Bn, int D)
{
    __shared__ __align__(16) _Float16 lA[2][128 * LDST];
    __shared__ __align__(16) _Float16 lB[2][128 * LDST];

    int tid  = threadIdx.x;
    int wave = tid >> 5;
    int lane = tid & 31;
    int mn   = lane & 15;    // M for A-frag / N for B-frag / N for D
    int kh   = lane >> 4;    // K-half selector
    int wM   = wave & 3;     // 4 row groups of 32 rows
    int wN   = wave >> 2;    // 2 col groups of 64 cols

    int rowBase = blockIdx.y * 128;
    int colBase = blockIdx.x * 128;

    v8f acc[2][4];
    #pragma unroll
    for (int g = 0; g < 2; ++g)
        #pragma unroll
        for (int nt = 0; nt < 4; ++nt)
            acc[g][nt] = (v8f){0.f,0.f,0.f,0.f,0.f,0.f,0.f,0.f};

    // ---- cooperative tile fetch: 128 rows x 32 halves = 512 chunks of 16B
    auto loadTile = [&](int buf, int k0) {
        #pragma unroll
        for (int c = tid; c < 512; c += 256) {
            int r   = c >> 2;
            int off = (c & 3) * 8;
            const _Float16* gA = &A [(size_t)(rowBase + r) * D + k0 + off];
            const _Float16* gB = &Bm[(size_t)(colBase + r) * D + k0 + off];
            _Float16* sA = &lA[buf][r * LDST + off];
            _Float16* sB = &lB[buf][r * LDST + off];
#ifdef HAVE_ASYNC_LDS
            __builtin_amdgcn_global_load_async_to_lds_b128(
                (glb_v4i*)gA, (lds_v4i*)sA, 0, 0);
            __builtin_amdgcn_global_load_async_to_lds_b128(
                (glb_v4i*)gB, (lds_v4i*)sB, 0, 0);
#else
            *(h8v*)sA = *(const h8v*)gA;
            *(h8v*)sB = *(const h8v*)gB;
#endif
        }
    };

    int nIter = D / 32;
    loadTile(0, 0);
    wait_async_all();
    __syncthreads();

    for (int it = 0; it < nIter; ++it) {
        int cur = it & 1;
        if (it + 1 < nIter) loadTile(1 - cur, (it + 1) * 32);   // prefetch next

        // A fragments for the wave's two 16-row groups
        HV16 af[2];
        #pragma unroll
        for (int g = 0; g < 2; ++g) {
            int arow = wM * 32 + g * 16 + mn;
            af[g].h[0] = *(const h8v*)&lA[cur][arow * LDST + kh * 8];
            af[g].h[1] = *(const h8v*)&lA[cur][arow * LDST + 16 + kh * 8];
        }
        #pragma unroll
        for (int nt = 0; nt < 4; ++nt) {
            HV16 bf;
            int brow = wN * 64 + nt * 16 + mn;
            bf.h[0] = *(const h8v*)&lB[cur][brow * LDST + kh * 8];
            bf.h[1] = *(const h8v*)&lB[cur][brow * LDST + 16 + kh * 8];
            #pragma unroll
            for (int g = 0; g < 2; ++g)
                acc[g][nt] = __builtin_amdgcn_wmma_f32_16x16x32_f16(
                    false, af[g].v, false, bf.v, (short)0, acc[g][nt], false, false);
        }

        wait_async_all();   // next tile landed (overlapped with WMMAs above)
        __syncthreads();    // everyone done reading `cur` before it's reused
    }

    // store D-matrix: VGPR r -> row = r + 8*kh, col = mn (+16*nt)
    #pragma unroll
    for (int g = 0; g < 2; ++g)
        #pragma unroll
        for (int nt = 0; nt < 4; ++nt)
            #pragma unroll
            for (int r = 0; r < 8; ++r) {
                int row = rowBase + wM * 32 + g * 16 + r + 8 * kh;
                int col = colBase + wN * 64 + nt * 16 + mn;
                C[(size_t)row * Bn + col] = acc[g][nt][r];
            }
}

// ---------------------------------------------------------------------------
// Kernel 3: image (row) side. One block per row i.
//   rowmax -> new_b_I -> exp sums -> per-row image loss.
// ---------------------------------------------------------------------------
__global__ __launch_bounds__(256) void row_pass(
    const float* __restrict__ sim, const float* __restrict__ s_I,
    const float* __restrict__ b_I, const int* __restrict__ ids,
    float* __restrict__ rowLoss, int Bn)
{
    int i = blockIdx.x;
    const float* r = sim + (size_t)i * Bn;
    __shared__ float red[256];
    __shared__ float sb[2];   // {old_b, new_b}
    int tid = threadIdx.x;
    float d = r[i];

    // pass 1: max of raw diffs (includes diag -> 0, matching reference)
    float mx = -3.402823466e38f;
    for (int j = tid; j < Bn; j += 256) mx = fmaxf(mx, r[j] - d);
    red[tid] = mx; __syncthreads();
    for (int s = 128; s > 0; s >>= 1) {
        if (tid < s) red[tid] = fmaxf(red[tid], red[tid + s]);
        __syncthreads();
    }
    if (tid == 0) {
        float ob = b_I[ids[i]];
        sb[0] = ob;
        sb[1] = fmaxf(red[0] * INV_T, ob);
    }
    __syncthreads();
    float nb = sb[1];

    // pass 2: sum exp and sum exp*diff (diag masked)
    float se = 0.0f, sw = 0.0f;
    for (int j = tid; j < Bn; j += 256) {
        if (j == i) continue;
        float diff = r[j] - d;
        float ex = __expf(fmaf(diff, INV_T, -nb));
        se += ex;
        sw = fmaf(ex, diff, sw);
    }
    red[tid] = se; __syncthreads();
    for (int s = 128; s > 0; s >>= 1) { if (tid < s) red[tid] += red[tid + s]; __syncthreads(); }
    float SE = red[0]; __syncthreads();
    red[tid] = sw; __syncthreads();
    for (int s = 128; s > 0; s >>= 1) { if (tid < s) red[tid] += red[tid + s]; __syncthreads(); }
    if (tid == 0) {
        float SW = red[0];
        float g  = SE / (float)(Bn - 1);
        float sn = (1.0f - GAMMA_C) * s_I[ids[i]] * __expf(sb[0] - nb) + GAMMA_C * g;
        float sc = fmaxf(sn, EPS_C);
        rowLoss[i] = SW / (sc * (float)(Bn - 1));
    }
}

// ---------------------------------------------------------------------------
// Kernel 4: text (column) side. One block per 64 columns; 256 threads laid out
// as (col = tid&63, rowgroup = tid>>6) so sim traversal is coalesced.
// ---------------------------------------------------------------------------
__global__ __launch_bounds__(256) void col_pass(
    const float* __restrict__ sim, const float* __restrict__ s_T,
    const float* __restrict__ b_T, const int* __restrict__ ids,
    float* __restrict__ colLoss, int Bn)
{
    __shared__ float red[256];
    __shared__ float sNew[64], sOld[64], sDiag[64];
    int tid = threadIdx.x;
    int c  = tid & 63;
    int rg = tid >> 6;
    int j  = blockIdx.x * 64 + c;

    if (rg == 0) sDiag[c] = sim[(size_t)j * Bn + j];
    __syncthreads();
    float d = sDiag[c];

    // pass 1: column max of raw diffs (includes diag -> 0)
    float mx = -3.402823466e38f;
    for (int i = rg; i < Bn; i += 4)
        mx = fmaxf(mx, sim[(size_t)i * Bn + j] - d);
    red[tid] = mx; __syncthreads();
    if (rg == 0) {
        float m = fmaxf(fmaxf(red[c], red[c + 64]), fmaxf(red[c + 128], red[c + 192]));
        float ob = b_T[ids[j]];
        sOld[c] = ob;
        sNew[c] = fmaxf(m * INV_T, ob);
    }
    __syncthreads();
    float nb = sNew[c];

    // pass 2: sums (diag masked)
    float se = 0.0f, sw = 0.0f;
    for (int i = rg; i < Bn; i += 4) {
        if (i == j) continue;
        float diff = sim[(size_t)i * Bn + j] - d;
        float ex = __expf(fmaf(diff, INV_T, -nb));
        se += ex;
        sw = fmaf(ex, diff, sw);
    }
    red[tid] = se; __syncthreads();
    float SE = 0.0f;
    if (rg == 0) SE = red[c] + red[c + 64] + red[c + 128] + red[c + 192];
    __syncthreads();
    red[tid] = sw; __syncthreads();
    if (rg == 0) {
        float SW = red[c] + red[c + 64] + red[c + 128] + red[c + 192];
        float g  = SE / (float)(Bn - 1);
        float sn = (1.0f - GAMMA_C) * s_T[ids[j]] * __expf(sOld[c] - nb) + GAMMA_C * g;
        float sc = fmaxf(sn, EPS_C);
        colLoss[j] = SW / (sc * (float)(Bn - 1));
    }
}

// ---------------------------------------------------------------------------
// Kernel 5: deterministic final reduction to the scalar loss.
// ---------------------------------------------------------------------------
__global__ __launch_bounds__(256) void final_reduce(
    const float* __restrict__ rowLoss, const float* __restrict__ colLoss,
    float* __restrict__ out, int Bn)
{
    __shared__ float red[256];
    int tid = threadIdx.x;
    float a = 0.0f, b = 0.0f;
    for (int i = tid; i < Bn; i += 256) { a += rowLoss[i]; b += colLoss[i]; }
    red[tid] = a; __syncthreads();
    for (int s = 128; s > 0; s >>= 1) { if (tid < s) red[tid] += red[tid + s]; __syncthreads(); }
    float RA = red[0]; __syncthreads();
    red[tid] = b; __syncthreads();
    for (int s = 128; s > 0; s >>= 1) { if (tid < s) red[tid] += red[tid + s]; __syncthreads(); }
    if (tid == 0) {
        float RB = red[0];
        out[0] = ALPHA_C * (RA / (float)Bn) + (1.0f - ALPHA_C) * (RB / (float)Bn);
    }
}

// ---------------------------------------------------------------------------
// Host-side launcher.
// Inputs (setup_inputs order): zis[B*D] f32, zjs[B*D] f32, s_I[N] f32,
// s_T[N] f32, b_I[N] f32, b_T[N] f32, ids[B] i32.  Output: scalar f32.
// Workspace: zn(f16 B*D) | wn(f16 B*D) | sim(f32 B*B) | rowLoss(B) | colLoss(B)
// ---------------------------------------------------------------------------
extern "C" void kernel_launch(void* const* d_in, const int* in_sizes, int n_in,
                              void* d_out, int out_size, void* d_ws, size_t ws_size,
                              hipStream_t stream)
{
    const float* zis = (const float*)d_in[0];
    const float* zjs = (const float*)d_in[1];
    const float* s_I = (const float*)d_in[2];
    const float* s_T = (const float*)d_in[3];
    const float* b_I = (const float*)d_in[4];
    const float* b_T = (const float*)d_in[5];
    const int*   ids = (const int*)  d_in[6];

    int Bn = in_sizes[6];            // 4096
    int D  = in_sizes[0] / Bn;       // 1024

    char* ws = (char*)d_ws;
    _Float16* zn = (_Float16*)ws;
    _Float16* wn = (_Float16*)(ws + (size_t)Bn * D * sizeof(_Float16));
    float* sim     = (float*)(ws + (size_t)2 * Bn * D * sizeof(_Float16));
    float* rowLoss = (float*)((char*)sim + (size_t)Bn * Bn * sizeof(float));
    float* colLoss = rowLoss + Bn;

    normalize_rows<<<2 * Bn, 256, 0, stream>>>(zis, zjs, zn, wn, Bn, D);

    dim3 grid(Bn / 128, Bn / 128);
    gemm_wmma<<<grid, 256, 0, stream>>>(zn, wn, sim, Bn, D);

    row_pass<<<Bn, 256, 0, stream>>>(sim, s_I, b_I, ids, rowLoss, Bn);
    col_pass<<<Bn / 64, 256, 0, stream>>>(sim, s_T, b_T, ids, colLoss, Bn);

    final_reduce<<<1, 256, 0, stream>>>(rowLoss, colLoss, (float*)d_out, Bn);
}
